// DSALVANet_38508676776068
// MI455X (gfx1250) — matrix operand order repaired and checked by reading, compiled, not run on hardware
//
#include <hip/hip_runtime.h>
#include <stdint.h>

// ---------------------------------------------------------------- constants
static constexpr int Bn = 2, Cn = 256, Hn = 96, Wn = 96, Pn = Hn * Wn; // 9216
static constexpr int OGn = 8, K2 = 9;
static constexpr int OCoff = 144;              // 2*OG*K2, multiple of 16
static constexpr int OCmsk = 72, OCmskPad = 80;
static constexpr int SIMC = 24, SIMCPad = 32;  // out-conv K padded to 32

#define HD __device__ __forceinline__

typedef __attribute__((ext_vector_type(16))) __bf16 v16b;
typedef __attribute__((ext_vector_type(16))) unsigned short v16u;
typedef __attribute__((ext_vector_type(8)))  float v8f;
typedef __attribute__((ext_vector_type(4)))  float v4f;
typedef __attribute__((ext_vector_type(4)))  unsigned int v4u;  // 16B
typedef __attribute__((ext_vector_type(8)))  unsigned int v8u;  // 32B

// fp32 -> bf16 round-to-nearest-even (bit trick)
HD unsigned short f2bf(float f) {
  unsigned int u = __builtin_bit_cast(unsigned int, f);
  u += 0x7FFFu + ((u >> 16) & 1u);
  return (unsigned short)(u >> 16);
}

// A fragment from row-major W[M][K] tile (row stride ld elements):
// lane m=lane&15, kb = (lane<16)?0:16 ; elems = K kb..kb+15
// -> two contiguous 16B chunks -> 2x global_load_b128
HD v16b loadA_frag(const unsigned short* A, int ld, int lane) {
  const unsigned short* r = A + (size_t)(lane & 15) * ld + ((lane < 16) ? 0 : 16);
  v4u lo = *(const v4u*)r;
  v4u hi = *(const v4u*)(r + 8);
  struct V { v4u a, b; } s{lo, hi};
  return __builtin_bit_cast(v16b, s);
}

// B fragment from channel-fastest activation: per-lane 32 contiguous bytes
HD v16b loadB32B(const unsigned short* p) {
  v8u v = *(const v8u*)p;
  return __builtin_bit_cast(v16b, v);
}
HD v16b zero_frag() { return __builtin_bit_cast(v16b, (v8u)0); }

// ---------------------------------------------------------------- converters
__global__ void k_f32_to_bf16(const float* __restrict__ in,
                              unsigned short* __restrict__ out, long n) {
  long i = (long)blockIdx.x * blockDim.x + threadIdx.x;
  if (i < n) out[i] = f2bf(in[i]);
}

// querys fp32 [B][256][P] -> xT bf16 [B][P][256] (channel fastest)
__global__ void k_cvt_xT(const float* __restrict__ in, unsigned short* __restrict__ out) {
  long i = (long)blockIdx.x * blockDim.x + threadIdx.x;
  if (i >= (long)Bn * Pn * Cn) return;
  long c = i % Cn, p = (i / Cn) % Pn, b = i / ((long)Cn * Pn);
  out[i] = f2bf(in[((size_t)b * Cn + c) * Pn + p]);
}

// out_w [256][24] -> bf16 [256][32] zero padded (row-major)
__global__ void k_cvt_outw(const float* __restrict__ w, unsigned short* __restrict__ o) {
  int i = blockIdx.x * blockDim.x + threadIdx.x;
  if (i >= 256 * SIMCPad) return;
  int oc = i / SIMCPad, k = i % SIMCPad;
  o[i] = (k < SIMC) ? f2bf(w[oc * SIMC + k]) : (unsigned short)0;
}

// conv3x3 weights [OC][256][9] -> bf16 [9][OCpad][256], zero-padded rows
__global__ void k_cvt_w3(const float* __restrict__ w, unsigned short* __restrict__ o,
                         int OC, int OCpad) {
  int i = blockIdx.x * blockDim.x + threadIdx.x;
  if (i >= 9 * OCpad * 256) return;
  int t = i / (OCpad * 256);
  int r = i % (OCpad * 256);
  int oc = r / 256, c = r % 256;
  o[i] = (oc < OC) ? f2bf(w[((size_t)oc * 256 + c) * 9 + t]) : (unsigned short)0;
}

// sim fp32 [B][P][24] -> bf16 [B][P][32] zero padded (both pixel-major)
__global__ void k_cvt_simT(const float* __restrict__ sim, unsigned short* __restrict__ o) {
  long i = (long)blockIdx.x * blockDim.x + threadIdx.x;
  if (i >= (long)Bn * Pn * SIMCPad) return;
  long r = i % SIMCPad, p = i / SIMCPad;   // p = b*Pn + pixel
  o[i] = (r < SIMC) ? f2bf(sim[p * SIMC + r]) : (unsigned short)0;
}

// ---------------------------------------------------------------- WMMA GEMM (1x1 conv)
// W [OC][Kd] row-major bf16, XT [B][P][Kd] bf16 (channel fastest)
// pixmajor!=0 : Out = [B][P][OC] (2x b128 stores/lane) ; else planar [B][OC][P]
__global__ void k_gemm1x1T(const unsigned short* __restrict__ W,
                           const unsigned short* __restrict__ XT,
                           const float* __restrict__ bias,
                           float* __restrict__ Out,
                           int OC, int Kd, int P, int pixmajor) {
  int lane = threadIdx.x & 31;
  int wave = threadIdx.x >> 5;
  int p0   = (blockIdx.x * (blockDim.x >> 5) + wave) * 16;
  int oc0  = blockIdx.y * 16;
  int b    = blockIdx.z;
  if (p0 >= P) return;
  int n  = lane & 15;
  int kh = (lane < 16) ? 0 : 16;
  const unsigned short* xp = XT + ((size_t)b * P + p0 + n) * Kd + kh;
  v8f acc = {};
  for (int k0 = 0; k0 < Kd; k0 += 32) {
    v16b a  = loadA_frag(W + (size_t)oc0 * Kd + k0, Kd, lane);
    v16b bb = loadB32B(xp + k0);
    acc = __builtin_amdgcn_wmma_f32_16x16x32_bf16(false, a, false, bb,
                                                  (short)0, acc, false, false);
  }
  int mo = (lane < 16) ? 0 : 8;
  int ocb = oc0 + mo;
  if (pixmajor) {
    float* Ob = Out + ((size_t)b * P + p0 + n) * OC + ocb;
    v4f lo = {acc[0] + bias[ocb + 0], acc[1] + bias[ocb + 1],
              acc[2] + bias[ocb + 2], acc[3] + bias[ocb + 3]};
    v4f hi = {acc[4] + bias[ocb + 4], acc[5] + bias[ocb + 5],
              acc[6] + bias[ocb + 6], acc[7] + bias[ocb + 7]};
    *(v4f*)Ob = lo;
    *(v4f*)(Ob + 4) = hi;
  } else {
    float* Ob = Out + ((size_t)b * OC + ocb) * P + p0 + n;
#pragma unroll
    for (int r = 0; r < 8; ++r) Ob[(size_t)r * P] = acc[r] + bias[ocb + r];
  }
}

// ---------------------------------------------------------------- WMMA 3x3 conv (pad=1, dil=1)
// Wt [9][OCpad][256] bf16, XT [B][P][256] bf16 channel-fastest,
// Out pixel-major [B][P][OC]
__global__ void k_conv3x3_wmma(const unsigned short* __restrict__ Wt,
                               const unsigned short* __restrict__ XT,
                               const float* __restrict__ bias,
                               float* __restrict__ Out,
                               int OC, int OCpad) {
  int lane = threadIdx.x & 31;
  int wave = threadIdx.x >> 5;
  int p0   = (blockIdx.x * (blockDim.x >> 5) + wave) * 16;
  if (p0 >= Pn) return;
  int y = p0 / Wn, x0 = p0 % Wn;   // W=96 multiple of 16: tile has fixed y
  int oc0 = blockIdx.y * 16;
  int b   = blockIdx.z;
  int n   = lane & 15;
  int kh  = (lane < 16) ? 0 : 16;
  const unsigned short* Xb = XT + (size_t)b * Pn * Cn;
  v8f acc = {};
#pragma unroll
  for (int t = 0; t < 9; ++t) {
    int dy = t / 3 - 1, dx = t % 3 - 1;
    int yy = y + dy;
    int xx = x0 + n + dx;
    bool ok = (yy >= 0) && (yy < Hn) && (xx >= 0) && (xx < Wn);
    const unsigned short* xp = Xb + ((size_t)yy * Wn + xx) * Cn + kh;
    const unsigned short* Wtt = Wt + ((size_t)t * OCpad + oc0) * Cn;
    for (int c0 = 0; c0 < Cn; c0 += 32) {
      v16b a  = loadA_frag(Wtt + c0, Cn, lane);
      v16b bb = ok ? loadB32B(xp + c0) : zero_frag();
      acc = __builtin_amdgcn_wmma_f32_16x16x32_bf16(false, a, false, bb,
                                                    (short)0, acc, false, false);
    }
  }
  int mo = (lane < 16) ? 0 : 8;
  int ocb = oc0 + mo;
  if (ocb < OC) {                      // OC multiple of 8 -> full-or-none per half
    float* Ob = Out + ((size_t)b * Pn + p0 + n) * OC + ocb;
    v4f lo = {acc[0] + bias[ocb + 0], acc[1] + bias[ocb + 1],
              acc[2] + bias[ocb + 2], acc[3] + bias[ocb + 3]};
    v4f hi = {acc[4] + bias[ocb + 4], acc[5] + bias[ocb + 5],
              acc[6] + bias[ocb + 6], acc[7] + bias[ocb + 7]};
    *(v4f*)Ob = lo;
    *(v4f*)(Ob + 4) = hi;
  }
}

// ---------------------------------------------------------------- LayerNorm over channels
// pre pixel-major [B][P][256] -> qf planar fp32 [B][256][P] + qT bf16 [B][P][256]
__global__ void k_ln(const float* __restrict__ pre, const float* __restrict__ g,
                     const float* __restrict__ bt, float* __restrict__ qf,
                     unsigned short* __restrict__ qT) {
  int lane = threadIdx.x & 31;
  int wave = threadIdx.x >> 5;
  int pix  = blockIdx.x * (blockDim.x >> 5) + wave;
  if (pix >= Bn * Pn) return;
  int b = pix / Pn, p = pix % Pn;
  const float* base = pre + (size_t)pix * Cn;      // contiguous per pixel
  float v[8], s = 0.f;
#pragma unroll
  for (int j = 0; j < 8; ++j) { v[j] = base[lane + j * 32]; s += v[j]; }
#pragma unroll
  for (int o = 16; o >= 1; o >>= 1) s += __shfl_xor(s, o, 32);
  float mu = s * (1.f / 256.f);
  float var = 0.f;
#pragma unroll
  for (int j = 0; j < 8; ++j) { float d = v[j] - mu; var += d * d; }
#pragma unroll
  for (int o = 16; o >= 1; o >>= 1) var += __shfl_xor(var, o, 32);
  float inv = rsqrtf(var * (1.f / 256.f) + 1e-5f);
  float* qo = qf + (size_t)b * Cn * Pn + p;
  unsigned short* qto = qT + (size_t)pix * Cn;
#pragma unroll
  for (int j = 0; j < 8; ++j) {
    int c = lane + j * 32;
    float r = (v[j] - mu) * inv * g[c] + bt[c];
    qo[(size_t)c * Pn] = r;          // planar for gather stage
    qto[c] = f2bf(r);                // channel-fastest for WMMA convs
  }
}

// ---------------------------------------------------------------- support path (tiny)
__global__ void k_support(const float* __restrict__ sup, const float* __restrict__ w,
                          const float* __restrict__ bias, const float* __restrict__ g,
                          const float* __restrict__ bt, float* __restrict__ sln) {
  __shared__ float buf[256];
  int pos = blockIdx.x;          // 0..53 = (b*3+e)*9+tap
  int be = pos / 9, tap = pos % 9;
  int oc = threadIdx.x;
  const float* sp = sup + (size_t)be * 256 * 9 + tap;
  const float* wr = w + (size_t)oc * 256;
  float acc = bias[oc];
  for (int c = 0; c < 256; ++c) acc += wr[c] * sp[(size_t)c * 9];
  buf[oc] = acc; __syncthreads();
  for (int s2 = 128; s2 > 0; s2 >>= 1) { if (oc < s2) buf[oc] += buf[oc + s2]; __syncthreads(); }
  float mu = buf[0] * (1.f / 256.f); __syncthreads();
  float d = acc - mu;
  buf[oc] = d * d; __syncthreads();
  for (int s2 = 128; s2 > 0; s2 >>= 1) { if (oc < s2) buf[oc] += buf[oc + s2]; __syncthreads(); }
  float var = buf[0] * (1.f / 256.f);
  float r = (acc - mu) * rsqrtf(var + 1e-5f) * g[oc] + bt[oc];
  sln[((size_t)be * 256 + oc) * 9 + tap] = r;
}

__global__ void k_pooled(const float* __restrict__ sln, float* __restrict__ pooled) {
  int i = blockIdx.x * blockDim.x + threadIdx.x;
  if (i >= Bn * 6 * 128) return;
  int b = i / (6 * 128), r = (i / 128) % 6, cg = i % 128;
  const float* sp = sln + (((size_t)(b * 3 + (r >> 1)) * 256) + ((r & 1) * 128 + cg)) * 9;
  float m = sp[0];
#pragma unroll
  for (int t = 1; t < 9; ++t) m = fmaxf(m, sp[t]);
  pooled[i] = m;
}

// ---------------------------------------------------------------- softmax over 72 chans, in-place
// msk pixel-major [B][P][72]: contiguous per pixel
__global__ void k_softmax72(float* __restrict__ m) {
  int idx = blockIdx.x * blockDim.x + threadIdx.x;
  if (idx >= Bn * Pn) return;
  float* mp = m + (size_t)idx * OCmsk;
  float mx = -3.4e38f;
  for (int c = 0; c < OCmsk; ++c) mx = fmaxf(mx, mp[c]);
  float s = 0.f;
  for (int c = 0; c < OCmsk; ++c) { float e = expf(mp[c] - mx); mp[c] = e; s += e; }
  float inv = 1.f / s;
  for (int c = 0; c < OCmsk; ++c) mp[c] *= inv;
}

// ---------------------------------------------------------------- sim0 (grouped 1x1 with pooled)
// sim pixel-major [B][P][24]
__global__ void k_sim0(const float* __restrict__ q, const float* __restrict__ pooled,
                       float* __restrict__ sim) {
  int idx = blockIdx.x * blockDim.x + threadIdx.x;
  if (idx >= Bn * Pn) return;
  int b = idx / Pn, p = idx % Pn;
  const float* qb = q + (size_t)b * Cn * Pn + p;
  const float* pl = pooled + (size_t)b * 6 * 128;
  float acc[6] = {0, 0, 0, 0, 0, 0};
  for (int g = 0; g < 2; ++g)
    for (int cg = 0; cg < 128; ++cg) {
      float qv = qb[(size_t)(g * 128 + cg) * Pn];
#pragma unroll
      for (int rr = 0; rr < 3; ++rr) acc[g * 3 + rr] += pl[(g * 3 + rr) * 128 + cg] * qv;
    }
  float* so = sim + (size_t)idx * SIMC;
#pragma unroll
  for (int r = 0; r < 6; ++r) so[r] = acc[r];
}

// ---------------------------------------------------------------- deformable conv (per dilation)
// One thread per (pixel, offset-group): 8x the parallelism of thread-per-pixel.
// og = lane&7 -> the 4 og's of a conv-group sit in one lane-quad; partials for the
// group's 3 output channels are combined with two shfl_xor steps.
__global__ void k_deform(const float* __restrict__ q,    // planar [B][256][P]
                         const float* __restrict__ off,  // pixel-major [B][P][144]
                         const float* __restrict__ msk,  // pixel-major [B][P][72]
                         const float* __restrict__ sln,  // [B][3][256][9]
                         float* __restrict__ sim,        // pixel-major [B][P][24]
                         int dil, int outc0) {
  int tid = blockIdx.x * blockDim.x + threadIdx.x;   // exact grid: Bn*Pn*8
  int og  = tid & 7;
  int idx = tid >> 3;                                // b*Pn + p
  int b = idx / Pn, p = idx % Pn;
  int y = p / Wn, x = p % Wn;
  int g = og >> 2;
  const float* qb = q + (size_t)b * Cn * Pn + (size_t)(og * 32) * Pn; // this og's 32 planes
  const float* ob = off + (size_t)idx * OCoff + og * 18;  // 18 contiguous floats
  const float* mb = msk + (size_t)idx * OCmsk + og * 9;   // 9 contiguous floats
  const float* sb = sln + (size_t)b * 3 * 256 * 9;
  // weight bases for the 3 output channels of this group, at this og's cg window
  const float* wb0; const float* wb1; const float* wb2;
  {
    int cgb = (og & 3) * 32;   // cg = cgb + cc
    int r0 = g * 3;
    wb0 = sb + ((size_t)((r0 + 0) >> 1) * 256 + ((r0 + 0) & 1) * 128 + cgb) * 9;
    wb1 = sb + ((size_t)((r0 + 1) >> 1) * 256 + ((r0 + 1) & 1) * 128 + cgb) * 9;
    wb2 = sb + ((size_t)((r0 + 2) >> 1) * 256 + ((r0 + 2) & 1) * 128 + cgb) * 9;
  }
  float a0 = 0.f, a1 = 0.f, a2 = 0.f;
#pragma unroll
  for (int t = 0; t < K2; ++t) {
    int ky = (t / 3) * dil - dil;
    int kx = (t % 3) * dil - dil;
    float dy = ob[t * 2 + 0];
    float dx = ob[t * 2 + 1];
    float mv = mb[t];
    float py = (float)(y + ky) + dy;
    float px = (float)(x + kx) + dx;
    float y0f = floorf(py), x0f = floorf(px);
    int y0 = (int)y0f, x0 = (int)x0f;
    float wy = py - y0f, wx = px - x0f;
    int y1 = y0 + 1, x1 = x0 + 1;
    bool iy0 = (y0 >= 0) & (y0 < Hn), iy1 = (y1 >= 0) & (y1 < Hn);
    bool ix0 = (x0 >= 0) & (x0 < Wn), ix1 = (x1 >= 0) & (x1 < Wn);
    int cy0 = min(max(y0, 0), Hn - 1), cy1 = min(max(y1, 0), Hn - 1);
    int cx0 = min(max(x0, 0), Wn - 1), cx1 = min(max(x1, 0), Wn - 1);
    int i00 = cy0 * Wn + cx0, i01 = cy0 * Wn + cx1;
    int i10 = cy1 * Wn + cx0, i11 = cy1 * Wn + cx1;
    float w00 = (iy0 && ix0) ? (1.f - wy) * (1.f - wx) * mv : 0.f;
    float w01 = (iy0 && ix1) ? (1.f - wy) * wx * mv : 0.f;
    float w10 = (iy1 && ix0) ? wy * (1.f - wx) * mv : 0.f;
    float w11 = (iy1 && ix1) ? wy * wx * mv : 0.f;
    __builtin_prefetch(qb + i00, 0, 3);               // global_prefetch
    for (int cc = 0; cc < 32; ++cc) {
      const float* img = qb + (size_t)cc * Pn;
      float sval = w00 * img[i00] + w01 * img[i01] + w10 * img[i10] + w11 * img[i11];
      a0 += wb0[cc * 9 + t] * sval;
      a1 += wb1[cc * 9 + t] * sval;
      a2 += wb2[cc * 9 + t] * sval;
    }
  }
  // reduce the 4 og's of this conv-group (lane-quad aligned since og = lane&7)
  a0 += __shfl_xor(a0, 1, 32); a0 += __shfl_xor(a0, 2, 32);
  a1 += __shfl_xor(a1, 1, 32); a1 += __shfl_xor(a1, 2, 32);
  a2 += __shfl_xor(a2, 1, 32); a2 += __shfl_xor(a2, 2, 32);
  if ((og & 3) == 0) {
    float* so = sim + (size_t)idx * SIMC + outc0 + g * 3;
    so[0] = a0; so[1] = a1; so[2] = a2;
  }
}

// ---------------------------------------------------------------- launch
extern "C" void kernel_launch(void* const* d_in, const int* in_sizes, int n_in,
                              void* d_out, int out_size, void* d_ws, size_t ws_size,
                              hipStream_t stream) {
  const float* querys = (const float*)d_in[0];
  const float* sup    = (const float*)d_in[1];
  const float* in_w   = (const float*)d_in[2];
  const float* in_b   = (const float*)d_in[3];
  const float* ln_g   = (const float*)d_in[4];
  const float* ln_b   = (const float*)d_in[5];
  const float* out_w  = (const float*)d_in[6];
  const float* out_b  = (const float*)d_in[7];
  const float* offw[3] = {(const float*)d_in[8],  (const float*)d_in[12], (const float*)d_in[16]};
  const float* offb[3] = {(const float*)d_in[9],  (const float*)d_in[13], (const float*)d_in[17]};
  const float* mskw[3] = {(const float*)d_in[10], (const float*)d_in[14], (const float*)d_in[18]};
  const float* mskb[3] = {(const float*)d_in[11], (const float*)d_in[15], (const float*)d_in[19]};

  // ---- workspace carve (256B aligned)
  char* ws = (char*)d_ws;
  size_t o = 0;
  auto carve = [&](size_t bytes) -> char* {
    char* p = ws + o;
    o += (bytes + 255) & ~(size_t)255;
    return p;
  };
  const long NQ = (long)Bn * Cn * Pn;                          // 4,718,592
  unsigned short* xT    = (unsigned short*)carve(NQ * 2);      // [B][P][256] bf16
  unsigned short* wbf   = (unsigned short*)carve(256 * 256 * 2);
  unsigned short* woutb = (unsigned short*)carve(256 * SIMCPad * 2);
  unsigned short* wtoff[3], *wtmsk[3];
  for (int d = 0; d < 3; ++d) wtoff[d] = (unsigned short*)carve((size_t)9 * OCoff * 256 * 2);
  for (int d = 0; d < 3; ++d) wtmsk[d] = (unsigned short*)carve((size_t)9 * OCmskPad * 256 * 2);
  float* qpreP = (float*)carve(NQ * 4);                        // [B][P][256]
  float* qf    = (float*)carve(NQ * 4);                        // [B][256][P]
  unsigned short* qT = (unsigned short*)carve(NQ * 2);         // [B][P][256]
  float* sln    = (float*)carve((size_t)Bn * 3 * 256 * 9 * 4);
  float* pooled = (float*)carve((size_t)Bn * 6 * 128 * 4);
  float* offo[3], *msko[3];
  for (int d = 0; d < 3; ++d) offo[d] = (float*)carve((size_t)Bn * Pn * OCoff * 4);
  for (int d = 0; d < 3; ++d) msko[d] = (float*)carve((size_t)Bn * Pn * OCmsk * 4);
  float* sim = (float*)carve((size_t)Bn * Pn * SIMC * 4);      // [B][P][24]
  unsigned short* simT = (unsigned short*)carve((size_t)Bn * Pn * SIMCPad * 2);
  (void)ws_size; (void)in_sizes; (void)n_in; (void)out_size;

  // ---- 1. conversions
  k_cvt_xT<<<(unsigned)((NQ + 255) / 256), 256, 0, stream>>>(querys, xT);
  k_f32_to_bf16<<<256, 256, 0, stream>>>(in_w, wbf, 256 * 256);
  k_cvt_outw<<<(256 * SIMCPad + 255) / 256, 256, 0, stream>>>(out_w, woutb);
  for (int d = 0; d < 3; ++d) {
    k_cvt_w3<<<(9 * OCoff * 256 + 255) / 256, 256, 0, stream>>>(offw[d], wtoff[d], OCoff, OCoff);
    k_cvt_w3<<<(9 * OCmskPad * 256 + 255) / 256, 256, 0, stream>>>(mskw[d], wtmsk[d], OCmsk, OCmskPad);
  }

  // ---- 2. q = LN(in_w * x + in_b)
  {
    dim3 g(Pn / 64, Cn / 16, Bn);
    k_gemm1x1T<<<g, 128, 0, stream>>>(wbf, xT, in_b, qpreP, Cn, Cn, Pn, 1);
  }
  k_ln<<<(Bn * Pn) / 4, 128, 0, stream>>>(qpreP, ln_g, ln_b, qf, qT);

  // ---- 3. support path + pooled
  k_support<<<Bn * 3 * 9, 256, 0, stream>>>(sup, in_w, in_b, ln_g, ln_b, sln);
  k_pooled<<<(Bn * 6 * 128 + 255) / 256, 256, 0, stream>>>(sln, pooled);

  // ---- 4. offset / mask 3x3 convs (WMMA) + softmax
  for (int d = 0; d < 3; ++d) {
    dim3 go(Pn / 64, OCoff / 16, Bn);
    k_conv3x3_wmma<<<go, 128, 0, stream>>>(wtoff[d], qT, offb[d], offo[d], OCoff, OCoff);
    dim3 gm(Pn / 64, OCmskPad / 16, Bn);
    k_conv3x3_wmma<<<gm, 128, 0, stream>>>(wtmsk[d], qT, mskb[d], msko[d], OCmsk, OCmskPad);
    k_softmax72<<<(Bn * Pn + 255) / 256, 256, 0, stream>>>(msko[d]);
  }

  // ---- 5. sim0 + deformable convs (dilation 1,2,4), 8 threads per pixel
  k_sim0<<<(Bn * Pn + 255) / 256, 256, 0, stream>>>(qf, pooled, sim);
  const int dils[3] = {1, 2, 4};
  for (int d = 0; d < 3; ++d)
    k_deform<<<(Bn * Pn * 8) / 256, 256, 0, stream>>>(qf, offo[d], msko[d], sln, sim,
                                                      dils[d], 6 + 6 * d);

  // ---- 6. out conv: sim -> bf16 padded, WMMA GEMM 32 -> 256, planar d_out
  {
    long tot = (long)Bn * Pn * SIMCPad;
    k_cvt_simT<<<(unsigned)((tot + 255) / 256), 256, 0, stream>>>(sim, simT);
    dim3 g(Pn / 64, Cn / 16, Bn);
    k_gemm1x1T<<<g, 128, 0, stream>>>(woutb, simT, out_b, (float*)d_out,
                                      Cn, SIMCPad, Pn, 0);
  }
}
